// SINKAttention_84198538870971
// MI455X (gfx1250) — matrix operand order repaired and checked by reading, not compiled, not run on hardware
//
#include <hip/hip_runtime.h>
#include <stdint.h>

#define DEV __device__ __forceinline__

typedef __bf16 bf16;
typedef __attribute__((ext_vector_type(16))) __bf16 v16bf;
typedef __attribute__((ext_vector_type(8)))  __bf16 v8bf;
typedef __attribute__((ext_vector_type(4)))  __bf16 v4bf;
typedef __attribute__((ext_vector_type(8)))  float  v8f;

constexpr int Bn = 4, Sn = 1024, Dn = 1024, Hn = 16, HDn = 64;
constexpr float LOGMU = -6.9314718055994531f;   // -ln(1024), also -ln(m)
constexpr float NEG_BIG = -1.0e9f;

// ---------------------------------------------------------------- WMMA helpers
DEV v8f wmma_bf16(v16bf a, v16bf b, v8f c) {
  return __builtin_amdgcn_wmma_f32_16x16x32_bf16(false, a, false, b, (short)0, c,
                                                 false, false);
}

// A fragment (16x32 bf16): row = lane&15, 16 contiguous K at k0+hi*16
DEV v16bf load_fragA_bf16(const bf16* __restrict__ base, int ld, int k0, int lane) {
  const int r = lane & 15, hi = lane >> 4;
  const bf16* p = base + (size_t)r * ld + k0 + hi * 16;
  v8bf lo = *(const v8bf*)p;
  v8bf hh = *(const v8bf*)(p + 8);
  v16bf f;
#pragma unroll
  for (int i = 0; i < 8; ++i) { f[i] = lo[i]; f[i + 8] = hh[i]; }
  return f;
}

// A fragment from fp32 source, converted to bf16 on the fly
DEV v16bf load_fragA_f32(const float* __restrict__ base, int ld, int k0, int lane) {
  const int r = lane & 15, hi = lane >> 4;
  const float* p = base + (size_t)r * ld + k0 + hi * 16;
  v8f lo = *(const v8f*)p;
  v8f hh = *(const v8f*)(p + 8);
  v16bf f;
#pragma unroll
  for (int i = 0; i < 8; ++i) { f[i] = (__bf16)lo[i]; f[i + 8] = (__bf16)hh[i]; }
  return f;
}

// B fragment (32x16 bf16): column = lane&15, 16 contiguous K at k0+khi*16
DEV v16bf load_fragB_bf16(const bf16* __restrict__ base, int ld, int k0, int lane) {
  const int n = lane & 15, khi = lane >> 4;
  return *(const v16bf*)(base + (size_t)n * ld + k0 + khi * 16);
}

// ------------------------------------------- async global -> LDS (16B / lane)
// VDST VGPR carries the LDS byte address (low 32 bits of generic LDS pointer);
// tracked by ASYNCcnt, waited with s_wait_asynccnt.
DEV void async_ld16(bf16* lds_dst, const bf16* gsrc) {
  unsigned loff = (unsigned)(uintptr_t)lds_dst;          // LDS aperture: addr[31:0]
  unsigned long long ga = (unsigned long long)(uintptr_t)gsrc;
  asm volatile("global_load_async_to_lds_b128 %0, %1, off"
               :: "v"(loff), "v"(ga) : "memory");
}
DEV void wait_async0() { asm volatile("s_wait_asynccnt 0x0" ::: "memory"); }

// ---------------------------------------------------------------- fp32 -> bf16
__global__ __launch_bounds__(256) void cvt_kernel(const float* __restrict__ in,
                                                  bf16* __restrict__ out, int n4) {
  int i = blockIdx.x * 256 + threadIdx.x;
  if (i >= n4) return;
  float4 f = ((const float4*)in)[i];
  v4bf o;
  o[0] = (__bf16)f.x; o[1] = (__bf16)f.y; o[2] = (__bf16)f.z; o[3] = (__bf16)f.w;
  ((v4bf*)out)[i] = o;
}

// -------------------------------------------------- X @ W^T + bias (Linear)
// LDS double-buffered via async-to-LDS; block tile 64(M) x 128(N), K stages of 32.
// MODE 0: bf16 out, head layout [B,H,S,HD]
// MODE 1: bf16 out, transposed head layout [B,H,HD,S]   (for V -> ctx B-operand)
// MODE 2: fp32 out, flat [B*S, D]                        (final projection)
template <int MODE>
__global__ __launch_bounds__(256) void gemm_xw_kernel(
    const bf16* __restrict__ X, const bf16* __restrict__ W,
    const float* __restrict__ bias, void* __restrict__ outp) {
  __shared__ bf16 smA[2][64 * 32];    // 4 KB per stage
  __shared__ bf16 smB[2][128 * 32];   // 8 KB per stage
  const int t = threadIdx.x;
  const int lane = t & 31;
  const int wave = t >> 5;
  const int row0blk = blockIdx.x * 64;
  const int col0blk = blockIdx.y * 128;
  const int wrow = (wave & 1) * 32;    // wave tile within block tile
  const int wcol = (wave >> 1) * 32;

  // stage loader: A tile 64x32 (256 x 16B chunks), B tile 128x32 (512 chunks)
  auto stage = [&](int buf, int k0) {
    {
      const int r = t >> 2, p = t & 3;
      async_ld16(&smA[buf][r * 32 + p * 8],
                 X + (size_t)(row0blk + r) * Dn + k0 + p * 8);
    }
#pragma unroll
    for (int q = 0; q < 2; ++q) {
      const int idx = t + q * 256;
      const int r = idx >> 2, p = idx & 3;
      async_ld16(&smB[buf][r * 32 + p * 8],
                 W + (size_t)(col0blk + r) * Dn + k0 + p * 8);
    }
  };

  stage(0, 0);
  v8f c00 = {}, c01 = {}, c10 = {}, c11 = {};
  constexpr int NST = Dn / 32;
  for (int s = 0; s < NST; ++s) {
    wait_async0();        // this wave's stage-s chunks are in LDS
    __syncthreads();      // everyone's chunks are in LDS; prior readers done
    if (s + 1 < NST) stage((s + 1) & 1, (s + 1) * 32);   // prefetch next stage
    const bf16* A = smA[s & 1];
    const bf16* Bt = smB[s & 1];
    // A frag: row = wrow + (lane&15), 16 contiguous K at hi*16
    const int r = lane & 15, hi = lane >> 4, khi = lane >> 4;
    v16bf a0, a1;
    {
      const bf16* p0 = A + (wrow + r) * 32 + hi * 16;
      const bf16* p1 = A + (wrow + 16 + r) * 32 + hi * 16;
      v8bf l0 = *(const v8bf*)p0, h0 = *(const v8bf*)(p0 + 8);
      v8bf l1 = *(const v8bf*)p1, h1 = *(const v8bf*)(p1 + 8);
#pragma unroll
      for (int i = 0; i < 8; ++i) {
        a0[i] = l0[i]; a0[i + 8] = h0[i];
        a1[i] = l1[i]; a1[i + 8] = h1[i];
      }
    }
    v16bf b0 = *(const v16bf*)(Bt + (wcol + r) * 32 + khi * 16);
    v16bf b1 = *(const v16bf*)(Bt + (wcol + 16 + r) * 32 + khi * 16);
    c00 = wmma_bf16(a0, b0, c00);
    c01 = wmma_bf16(a0, b1, c01);
    c10 = wmma_bf16(a1, b0, c10);
    c11 = wmma_bf16(a1, b1, c11);
  }

  const int nlo = lane & 15, hi = lane >> 4;
  const int row0 = row0blk + wrow;
  const int col0 = col0blk + wcol;
#pragma unroll
  for (int i = 0; i < 2; ++i) {
#pragma unroll
    for (int j = 0; j < 2; ++j) {
      v8f acc = (i == 0) ? (j == 0 ? c00 : c01) : (j == 0 ? c10 : c11);
#pragma unroll
      for (int r = 0; r < 8; ++r) {
        const int m = row0 + i * 16 + hi * 8 + r;
        const int n = col0 + j * 16 + nlo;
        const float val = acc[r] + bias[n];
        if (MODE == 2) {
          ((float*)outp)[(size_t)m * Dn + n] = val;
        } else {
          const int b = m >> 10, s2 = m & (Sn - 1);
          const int h = n >> 6, hd = n & (HDn - 1);
          if (MODE == 0)
            ((bf16*)outp)[(((size_t)b * Hn + h) * Sn + s2) * HDn + hd] = (__bf16)val;
          else
            ((bf16*)outp)[(((size_t)b * Hn + h) * HDn + hd) * Sn + s2] = (__bf16)val;
        }
      }
    }
  }
}

// ---------------------------------------------- scores = q k^T / 8, causal mask
__global__ __launch_bounds__(256) void scores_kernel(const bf16* __restrict__ qh,
                                                     const bf16* __restrict__ kh,
                                                     float* __restrict__ attn) {
  const int lane = threadIdx.x & 31;
  const int wave = threadIdx.x >> 5;
  const int slice = blockIdx.z;                       // b*H + h
  const int q0 = blockIdx.x * 64 + (wave & 1) * 32;
  const int n0 = blockIdx.y * 128 + (wave >> 1) * 32; // key index base
  float* as = attn + ((size_t)slice << 20);
  const int nlo = lane & 15, hi = lane >> 4;

  if (n0 > q0) {  // wave tile entirely above diagonal -> all masked
#pragma unroll
    for (int i = 0; i < 2; ++i)
#pragma unroll
      for (int j = 0; j < 2; ++j)
#pragma unroll
        for (int r = 0; r < 8; ++r)
          as[(size_t)(q0 + i * 16 + hi * 8 + r) * Sn + (n0 + j * 16 + nlo)] = NEG_BIG;
    return;
  }

  const bf16* qs = qh + ((size_t)slice << 16);  // S*HD
  const bf16* ks = kh + ((size_t)slice << 16);
  v8f c00 = {}, c01 = {}, c10 = {}, c11 = {};
#pragma unroll
  for (int k0 = 0; k0 < HDn; k0 += 32) {
    v16bf a0 = load_fragA_bf16(qs + (size_t)q0 * HDn, HDn, k0, lane);
    v16bf a1 = load_fragA_bf16(qs + (size_t)(q0 + 16) * HDn, HDn, k0, lane);
    v16bf b0 = load_fragB_bf16(ks + (size_t)n0 * HDn, HDn, k0, lane);
    v16bf b1 = load_fragB_bf16(ks + (size_t)(n0 + 16) * HDn, HDn, k0, lane);
    c00 = wmma_bf16(a0, b0, c00);
    c01 = wmma_bf16(a0, b1, c01);
    c10 = wmma_bf16(a1, b0, c10);
    c11 = wmma_bf16(a1, b1, c11);
  }
#pragma unroll
  for (int i = 0; i < 2; ++i)
#pragma unroll
    for (int j = 0; j < 2; ++j) {
      v8f acc = (i == 0) ? (j == 0 ? c00 : c01) : (j == 0 ? c10 : c11);
#pragma unroll
      for (int r = 0; r < 8; ++r) {
        const int m = q0 + i * 16 + hi * 8 + r;
        const int n = n0 + j * 16 + nlo;
        float val = acc[r] * 0.125f;                  // 1/sqrt(64)
        if (n > m) val = NEG_BIG;
        as[(size_t)m * Sn + n] = val;
      }
    }
}

// ------------------------------------------------- Sinkhorn row pass: update u
// u[q] = log_mu - LSE_k( S[q,k] + v[k] )     (one wave per row, streaming LSE)
__global__ __launch_bounds__(256) void row_lse_kernel(const float* __restrict__ attn,
                                                      const float* __restrict__ vbuf,
                                                      float* __restrict__ ubuf) {
  const int lane = threadIdx.x & 31;
  const int wave = threadIdx.x >> 5;
  const int row = blockIdx.x * 8 + wave;      // 0 .. B*H*S-1
  const int slice = row >> 10, q = row & (Sn - 1);
  const float* Srow = attn + ((size_t)slice << 20) + ((size_t)q << 10);
  const float* vv = vbuf + (slice << 10);
  float m = -INFINITY, s = 0.f;
  for (int k = lane; k < Sn; k += 32) {
    float x = Srow[k] + vv[k];
    float nm = fmaxf(m, x);
    s = s * __expf(m - nm) + __expf(x - nm);
    m = nm;
  }
#pragma unroll
  for (int off = 16; off; off >>= 1) {
    float om = __shfl_xor(m, off, 32);
    float os = __shfl_xor(s, off, 32);
    float nm = fmaxf(m, om);
    s = s * __expf(m - nm) + os * __expf(om - nm);
    m = nm;
  }
  if (lane == 0) ubuf[row] = LOGMU - (m + __logf(s));
}

// ------------------------------------------------- Sinkhorn col pass: update v
// v[k] = log_nu - LSE_q( S[q,k] + u[q] )     (thread per column, coalesced rows)
__global__ __launch_bounds__(256) void col_lse_kernel(const float* __restrict__ attn,
                                                      const float* __restrict__ ubuf,
                                                      float* __restrict__ vbuf) {
  const int c = blockIdx.x * 256 + threadIdx.x;
  const int slice = blockIdx.y;
  const float* Sb = attn + ((size_t)slice << 20);
  const float* uu = ubuf + (slice << 10);
  float m = -INFINITY, s = 0.f;
  for (int q = 0; q < Sn; ++q) {
    float x = Sb[((size_t)q << 10) + c] + uu[q];
    float nm = fmaxf(m, x);
    s = s * __expf(m - nm) + __expf(x - nm);
    m = nm;
  }
  vbuf[(slice << 10) + c] = LOGMU - (m + __logf(s));
}

// -------------------------------- pi = exp(S + u + v) * m, in place over attn
__global__ __launch_bounds__(256) void pi_kernel(float* __restrict__ attn,
                                                 const float* __restrict__ ubuf,
                                                 const float* __restrict__ vbuf) {
  const size_t base = ((size_t)blockIdx.x * 256 + threadIdx.x) * 4;
  const int slice = (int)(base >> 20);
  const int q = (int)(base >> 10) & (Sn - 1);
  const int k = (int)base & (Sn - 1);
  const float u = ubuf[(slice << 10) + q];
  const float4 vv = *(const float4*)(vbuf + (slice << 10) + k);
  float4 x = *(float4*)(attn + base);
  x.x = __expf(x.x + u + vv.x) * 1024.f;
  x.y = __expf(x.y + u + vv.y) * 1024.f;
  x.z = __expf(x.z + u + vv.z) * 1024.f;
  x.w = __expf(x.w + u + vv.w) * 1024.f;
  *(float4*)(attn + base) = x;
}

// -------------------------------------------- ctx = pi @ v  (per (b,h) slice)
__global__ __launch_bounds__(256) void ctx_kernel(const float* __restrict__ attn,
                                                  const bf16* __restrict__ vt,
                                                  bf16* __restrict__ ctxbf) {
  const int lane = threadIdx.x & 31;
  const int wave = threadIdx.x >> 5;
  const int slice = blockIdx.z;
  const int q0 = blockIdx.x * 128 + (wave >> 1) * 32;
  const int n0 = (wave & 1) * 32;                 // over HD = 64
  const float* as = attn + ((size_t)slice << 20);
  const bf16* vs = vt + ((size_t)slice << 16);    // [HD, S] K-contiguous
  v8f c00 = {}, c01 = {}, c10 = {}, c11 = {};
  for (int k0 = 0; k0 <= q0; k0 += 32) {          // pi==0 above diagonal
    v16bf a0 = load_fragA_f32(as + (size_t)q0 * Sn, Sn, k0, lane);
    v16bf a1 = load_fragA_f32(as + (size_t)(q0 + 16) * Sn, Sn, k0, lane);
    v16bf b0 = load_fragB_bf16(vs + (size_t)n0 * Sn, Sn, k0, lane);
    v16bf b1 = load_fragB_bf16(vs + (size_t)(n0 + 16) * Sn, Sn, k0, lane);
    c00 = wmma_bf16(a0, b0, c00);
    c01 = wmma_bf16(a0, b1, c01);
    c10 = wmma_bf16(a1, b0, c10);
    c11 = wmma_bf16(a1, b1, c11);
  }
  const int b = slice >> 4, h = slice & (Hn - 1);
  const int nlo = lane & 15, hi = lane >> 4;
#pragma unroll
  for (int i = 0; i < 2; ++i)
#pragma unroll
    for (int j = 0; j < 2; ++j) {
      v8f acc = (i == 0) ? (j == 0 ? c00 : c01) : (j == 0 ? c10 : c11);
#pragma unroll
      for (int r = 0; r < 8; ++r) {
        const int q = q0 + i * 16 + hi * 8 + r;
        const int n = n0 + j * 16 + nlo;
        ctxbf[((size_t)b * Sn + q) * Dn + (h * HDn + n)] = (__bf16)acc[r];
      }
    }
}

// ============================================================== host launcher
extern "C" void kernel_launch(void* const* d_in, const int* in_sizes, int n_in,
                              void* d_out, int out_size, void* d_ws, size_t ws_size,
                              hipStream_t stream) {
  const float* Q   = (const float*)d_in[0];
  const float* K   = (const float*)d_in[1];
  const float* V   = (const float*)d_in[2];
  // d_in[3] = attn_mask (causal triu) — implemented analytically in-kernel
  const float* WQw = (const float*)d_in[4];
  const float* WQb = (const float*)d_in[5];
  const float* WKw = (const float*)d_in[6];
  const float* WKb = (const float*)d_in[7];
  const float* WVw = (const float*)d_in[8];
  const float* WVb = (const float*)d_in[9];
  const float* WOw = (const float*)d_in[10];
  const float* WOb = (const float*)d_in[11];

  float* out  = (float*)d_out;                       // [B,S,D]
  float* attn = out + (size_t)Bn * Sn * Dn;          // [B,H,S,S] (also scratch)

  // workspace carve-up (~64.5 MB)
  char* w = (char*)d_ws;
  size_t off = 0;
  auto carve = [&](size_t bytes) { char* p = w + off; off += bytes; return p; };
  const size_t XB = (size_t)Bn * Sn * Dn * sizeof(bf16);   // 8 MB
  const size_t WB = (size_t)Dn * Dn * sizeof(bf16);        // 2 MB
  bf16* Qbf   = (bf16*)carve(XB);
  bf16* Kbf   = (bf16*)carve(XB);
  bf16* Vbf   = (bf16*)carve(XB);
  bf16* WQbf  = (bf16*)carve(WB);
  bf16* WKbf  = (bf16*)carve(WB);
  bf16* WVbf  = (bf16*)carve(WB);
  bf16* WObf  = (bf16*)carve(WB);
  bf16* qh    = (bf16*)carve(XB);    // [B,H,S,HD]
  bf16* kh    = (bf16*)carve(XB);    // [B,H,S,HD]
  bf16* vt    = (bf16*)carve(XB);    // [B,H,HD,S]
  bf16* ctxbf = (bf16*)carve(XB);    // [B,S,D]
  float* ubuf = (float*)carve((size_t)Bn * Hn * Sn * sizeof(float));
  float* vbuf = (float*)carve((size_t)Bn * Hn * Sn * sizeof(float));

  const int NX4 = Bn * Sn * Dn / 4;   // 1,048,576
  const int NW4 = Dn * Dn / 4;        //   262,144
  cvt_kernel<<<NX4 / 256, 256, 0, stream>>>(Q, Qbf, NX4);
  cvt_kernel<<<NX4 / 256, 256, 0, stream>>>(K, Kbf, NX4);
  cvt_kernel<<<NX4 / 256, 256, 0, stream>>>(V, Vbf, NX4);
  cvt_kernel<<<NW4 / 256, 256, 0, stream>>>(WQw, WQbf, NW4);
  cvt_kernel<<<NW4 / 256, 256, 0, stream>>>(WKw, WKbf, NW4);
  cvt_kernel<<<NW4 / 256, 256, 0, stream>>>(WVw, WVbf, NW4);
  cvt_kernel<<<NW4 / 256, 256, 0, stream>>>(WOw, WObf, NW4);

  dim3 gp(Bn * Sn / 64, Dn / 128);                  // (64, 8)
  gemm_xw_kernel<0><<<gp, 256, 0, stream>>>(Qbf, WQbf, WQb, qh);
  gemm_xw_kernel<0><<<gp, 256, 0, stream>>>(Kbf, WKbf, WKb, kh);
  gemm_xw_kernel<1><<<gp, 256, 0, stream>>>(Vbf, WVbf, WVb, vt);

  dim3 gsc(Sn / 64, Sn / 128, Bn * Hn);             // (16, 8, 64)
  scores_kernel<<<gsc, 256, 0, stream>>>(qh, kh, attn);

  hipMemsetAsync(vbuf, 0, (size_t)Bn * Hn * Sn * sizeof(float), stream);
  for (int it = 0; it < 3; ++it) {
    row_lse_kernel<<<Bn * Hn * Sn / 8, 256, 0, stream>>>(attn, vbuf, ubuf);
    col_lse_kernel<<<dim3(Sn / 256, Bn * Hn), 256, 0, stream>>>(attn, ubuf, vbuf);
  }
  pi_kernel<<<(unsigned)((size_t)Bn * Hn * Sn * Sn / 1024), 256, 0, stream>>>(attn, ubuf, vbuf);

  ctx_kernel<<<dim3(Sn / 128, 1, Bn * Hn), 256, 0, stream>>>(attn, vt, ctxbf);

  gemm_xw_kernel<2><<<gp, 256, 0, stream>>>(ctxbf, WObf, WOb, out);
}
